// DGCN_HGN_AD_9543417332132
// MI455X (gfx1250) — matrix operand order repaired and checked by reading, not compiled, not run on hardware
//
#include <hip/hip_runtime.h>

typedef __bf16 bf16;
typedef __attribute__((ext_vector_type(16))) __bf16 v16bf;
typedef __attribute__((ext_vector_type(8)))  float  v8f;

#define BB    16
#define NNODE 1024
#define TT    16
#define FF    64
#define MEDGE 256
#define NHID  128
#define BN_   (BB*NNODE)        // 16384
#define BNT_  (BB*NNODE*TT)     // 262144

__device__ __forceinline__ float sigm(float x){ return 1.0f/(1.0f+__expf(-x)); }

// ---------------- WMMA fragment swizzle (bf16, wave32) ----------------
// A 16x32 fragment: lane=(r&15)+half*16, half=(k>>3)&1, elem e=((k>>4)&1)*8+(k&7)
// B 32x16 fragment: lane=(n&15)+half*16, half=(k>>3)&1, elem e=((k>>4)&1)*8+(k&7)
// storage: tile-major, 32 lanes x 16 bf16 (=1KB) per 16x32 / 32x16 tile
__device__ __forceinline__ long long aswz(int r, int k, int K){
    const int tm = r >> 4, tk = k >> 5, kp = k & 31;
    const int lane = (r & 15) + (((kp >> 3) & 1) << 4);
    const int e = (((kp >> 4) & 1) << 3) + (kp & 7);
    return (((long long)tm * (K >> 5) + tk) * 32 + lane) * 16 + e;
}
__device__ __forceinline__ long long bswz(int k, int n, int N){
    const int tk = k >> 5, tn = n >> 4;
    const int lane = (n & 15) + (((k >> 3) & 1) << 4);
    return (((long long)tk * (N >> 4) + tn) * 32 + lane) * 16 + ((((k >> 4) & 1) << 3) + (k & 7));
}

union FragU { v16bf v; uint4 q[2]; };

// ======================= generic batched WMMA bf16 GEMM =======================
// One wave computes a 32x64 macro-tile (2x4 WMMA tiles): per K=32 step,
// 2 A-fragment + 4 B-fragment b128-pair loads feed 8 v_wmma ops.
// A, B pre-swizzled bf16. Cf: optional f32 row-major out (also acc source).
// Cb: optional bf16 out, stored pre-swizzled per cbMode for the consumer GEMM:
//  1: A-swz direct, viewK=cbP1            2: B-swz direct, viewN=cbP1
//  3: A-swz, [b][R,T*w]->[R*T,w] reshape, w=1<<cbP2 (uses cbase)
//  4: B-swz, [BNT,w]->[b][1024,T*w], viewN=cbP1, w=1<<cbP2
//  5: B-swz, [BN,N]->[b][1024,N],   viewN=cbP1
__global__ __launch_bounds__(256)
void gemm_wmma_bf16(const bf16* __restrict__ A, long long sA,
                    const bf16* __restrict__ Bm, long long sB,
                    float* Cf, bf16* Cb, int ldc, long long sC,
                    const float* __restrict__ bias, int biasMask, float biasScale,
                    int M, int N, int K, int act, int acc,
                    int cbMode, int cbP1, int cbP2)
{
    const int lane = threadIdx.x & 31;
    const int wave = threadIdx.x >> 5;
    const int tm   = blockIdx.y * 8 + wave;      // 32-row macro-tile index (wave-uniform)
    if (tm * 32 >= M) return;                    // uniform exit: EXEC stays all-ones
    const int tn = blockIdx.x;                   // 64-col macro-tile index
    A  += (long long)blockIdx.z * sA;
    Bm += (long long)blockIdx.z * sB;
    const long long cbase = (long long)blockIdx.z * sC;
    const int r = lane & 15, half = lane >> 4;

    v8f c[2][4];
    if (acc) {
        #pragma unroll
        for (int mi = 0; mi < 2; ++mi)
            #pragma unroll
            for (int ni = 0; ni < 4; ++ni)
                #pragma unroll
                for (int i = 0; i < 8; ++i)
                    c[mi][ni][i] = Cf[cbase
                        + (long long)(tm*32 + mi*16 + half*8 + i)*ldc
                        + tn*64 + ni*16 + r];
    } else {
        #pragma unroll
        for (int mi = 0; mi < 2; ++mi)
            #pragma unroll
            for (int ni = 0; ni < 4; ++ni)
                #pragma unroll
                for (int i = 0; i < 8; ++i) c[mi][ni][i] = 0.0f;
    }

    const int ktiles = K >> 5;
    const bf16* Ap0 = A  + ((long long)(tm*2)     * ktiles * 32 + lane) * 16;
    const bf16* Ap1 = A  + ((long long)(tm*2 + 1) * ktiles * 32 + lane) * 16;
    const bf16* Bp  = Bm + ((long long)(tn*4) * 32 + lane) * 16;
    const long long bstep = (long long)(N >> 4) * 512;
    for (int kt = 0; kt < ktiles; ++kt) {
        FragU fa[2], fb[4];
        const uint4* a0 = (const uint4*)(Ap0 + (long long)kt * 512);
        fa[0].q[0] = a0[0]; fa[0].q[1] = a0[1];
        const uint4* a1 = (const uint4*)(Ap1 + (long long)kt * 512);
        fa[1].q[0] = a1[0]; fa[1].q[1] = a1[1];
        #pragma unroll
        for (int ni = 0; ni < 4; ++ni) {
            const uint4* bp = (const uint4*)(Bp + (long long)kt * bstep + ni * 512);
            fb[ni].q[0] = bp[0]; fb[ni].q[1] = bp[1];
        }
        #pragma unroll
        for (int mi = 0; mi < 2; ++mi)
            #pragma unroll
            for (int ni = 0; ni < 4; ++ni)
                c[mi][ni] = __builtin_amdgcn_wmma_f32_16x16x32_bf16(
                    false, fa[mi].v, false, fb[ni].v, (short)0, c[mi][ni], false, false);
    }

    #pragma unroll
    for (int ni = 0; ni < 4; ++ni) {
        const int col = tn*64 + ni*16 + r;
        const float bv = bias ? bias[col & biasMask] * biasScale : 0.0f;
        #pragma unroll
        for (int mi = 0; mi < 2; ++mi) {
            #pragma unroll
            for (int i = 0; i < 8; ++i) {
                float v = c[mi][ni][i] + bv;
                if (act == 1)      v = fmaxf(v, 0.0f);
                else if (act == 2) v = tanhf(v);
                const int row = tm*32 + mi*16 + half*8 + i;
                if (Cf) Cf[cbase + (long long)row*ldc + col] = v;
                if (Cb) {
                    long long idx;
                    if (cbMode == 1) {
                        idx = cbase + aswz(row, col, cbP1);
                    } else if (cbMode == 3) {
                        idx = cbase + aswz(row*TT + (col >> cbP2),
                                           col & ((1 << cbP2) - 1), 1 << cbP2);
                    } else if (cbMode == 4) {
                        const int bt = row >> 14, brow = (row >> 4) & 1023;
                        const int bn = ((row & 15) << cbP2) + col;
                        idx = (long long)bt * 1024 * cbP1 + bswz(brow, bn, cbP1);
                    } else if (cbMode == 5) {
                        const int bt = row >> 10;
                        idx = (long long)bt * 1024 * cbP1 + bswz(row & 1023, col, cbP1);
                    } else {
                        idx = cbase + bswz(row, col, cbP1);
                    }
                    Cb[idx] = (bf16)v;
                }
            }
        }
    }
}

// ======================= staging: f32 -> swizzled bf16 =======================
// logical L[R,C]: L[r][c] = trans ? src[c*R + r] : src[r*C + c]
// layoutB==0 -> A-swz(view=K), layoutB==1 -> B-swz(view=N total, column offset colOff)
__global__ void stage_k(const float* __restrict__ src, int R, int C, int trans,
                        int layoutB, bf16* __restrict__ dst, int view, int colOff)
{
    long long i = (long long)blockIdx.x * blockDim.x + threadIdx.x;
    if (i >= (long long)R * C) return;
    const int r = (int)(i / C), c = (int)(i % C);
    const float v = trans ? src[(long long)c * R + r] : src[i];
    const long long idx = layoutB ? bswz(r, c + colOff, view) : aswz(r, c, view);
    dst[idx] = (bf16)v;
}

// ======================= GRU gate =======================
__global__ void gru_gate_k(const float* __restrict__ gx, const float* __restrict__ gh,
                           const float* __restrict__ bhh, float* __restrict__ h,
                           bf16* __restrict__ hswz, bf16* __restrict__ outb,
                           float* __restrict__ outf, int t, int n)
{
    int i = blockIdx.x * blockDim.x + threadIdx.x;
    if (i >= n) return;
    const int b = i >> 7, j = i & 127;
    const long long gxo = ((long long)b * TT + t) * 384;
    float gr, gz, gn, hp;
    if (gh) {
        const float* g = gh + (long long)b * 384;
        gr = g[j]; gz = g[128 + j]; gn = g[256 + j];
        hp = h[i];
    } else {
        gr = bhh[j]; gz = bhh[128 + j]; gn = bhh[256 + j];
        hp = 0.0f;
    }
    const float rr = sigm(gx[gxo + j] + gr);
    const float zz = sigm(gx[gxo + 128 + j] + gz);
    const float nn = tanhf(gx[gxo + 256 + j] + rr * gn);
    const float hv = (1.0f - zz) * nn + zz * hp;
    h[i] = hv;
    const bf16 hb = (bf16)hv;
    hswz[aswz(b, j, 128)] = hb;                       // recurrent A operand [BN,128]
    outb[aswz(b * TT + t, j, 128)] = hb;              // sequence A operand [BNT,128]
    if (outf) outf[((long long)b * TT + t) * 128 + j] = hv;
}

// ======================= causal MHA core =======================
__global__ void attn_core_k(const float* __restrict__ qkv, bf16* __restrict__ ob, int total)
{
    int i = blockIdx.x * blockDim.x + threadIdx.x;
    if (i >= total) return;
    const int tq = i & 15, hh = (i >> 4) & 3, bn = i >> 6;
    const float* base = qkv + (long long)bn * TT * 384;
    const float* q = base + tq * 384 + hh * 32;
    float s[16];
    float mx = -3.0e38f;
    for (int tk = 0; tk <= tq; ++tk) {
        const float* k = base + tk * 384 + 128 + hh * 32;
        float acc = 0.0f;
        for (int d = 0; d < 32; ++d) acc += q[d] * k[d];
        acc *= 0.17677669529663687f;          // 1/sqrt(32)
        s[tk] = acc;
        mx = fmaxf(mx, acc);
    }
    float sum = 0.0f;
    for (int tk = 0; tk <= tq; ++tk) { s[tk] = __expf(s[tk] - mx); sum += s[tk]; }
    const float inv = 1.0f / sum;
    const int row = bn * TT + tq;
    for (int d = 0; d < 32; ++d) {
        float acc = 0.0f;
        for (int tk = 0; tk <= tq; ++tk) acc += s[tk] * base[tk * 384 + 256 + hh * 32 + d];
        ob[aswz(row, hh * 32 + d, 128)] = (bf16)(acc * inv);
    }
}

// ======================= residual + layernorm -> swizzled enc =======================
__global__ void ln_res_k(const float* __restrict__ a, const float* __restrict__ b,
                         const float* __restrict__ g, const float* __restrict__ beta,
                         bf16* __restrict__ out, int rows)
{
    int r = blockIdx.x * blockDim.x + threadIdx.x;
    if (r >= rows) return;
    const float* x = a + (long long)r * 128;
    const float* y = b + (long long)r * 128;
    float s = 0.0f, sq = 0.0f;
    for (int j = 0; j < 128; ++j) { float v = x[j] + y[j]; s += v; sq += v * v; }
    const float mu  = s * (1.0f / 128.0f);
    const float var = sq * (1.0f / 128.0f) - mu * mu;
    const float inv = rsqrtf(var + 1e-6f);
    const int bn = r >> 4, cb = (r & 15) * 128;        // enc view: [BN, T*128]
    for (int j = 0; j < 128; ++j) {
        float v = x[j] + y[j];
        out[aswz(bn, cb + j, 2048)] = (bf16)(((v - mu) * inv) * g[j] + beta[j]);
    }
}

// ======================= small tail kernels =======================
__global__ void rowdot_k(const float* __restrict__ a, const float* __restrict__ u,
                         float* __restrict__ out, int n, int k)
{
    int i = blockIdx.x * blockDim.x + threadIdx.x;
    if (i >= n) return;
    const float* r = a + (long long)i * k;
    float s = 0.0f;
    for (int j = 0; j < k; ++j) s += r[j] * u[j];
    out[i] = s;
}

__global__ void softmax_bn_k(const float* __restrict__ l, float* __restrict__ alpha, int n)
{
    __shared__ float red[256];
    const int tid = threadIdx.x;
    float mx = -3.0e38f;
    for (int i = tid; i < n; i += 256) mx = fmaxf(mx, l[i]);
    red[tid] = mx; __syncthreads();
    for (int s = 128; s > 0; s >>= 1) { if (tid < s) red[tid] = fmaxf(red[tid], red[tid + s]); __syncthreads(); }
    mx = red[0]; __syncthreads();
    float sum = 0.0f;
    for (int i = tid; i < n; i += 256) sum += __expf(l[i] - mx);
    red[tid] = sum; __syncthreads();
    for (int s = 128; s > 0; s >>= 1) { if (tid < s) red[tid] += red[tid + s]; __syncthreads(); }
    const float inv = 1.0f / red[0];
    for (int i = tid; i < n; i += 256) alpha[i] = __expf(l[i] - mx) * inv;
}

__global__ void acon_k(const float* __restrict__ hgn, const float* __restrict__ alpha,
                       float* __restrict__ acon, int n)
{
    int i = blockIdx.x * blockDim.x + threadIdx.x;
    if (i >= n) return;
    const float* h = hgn + (long long)i * 128;
    float s = 0.0f;
    for (int j = 0; j < 128; ++j) s += h[j];
    acon[i] = s * alpha[i];
}

__global__ void heads_k(const float* __restrict__ hgn, const float* __restrict__ acon,
                        const float* __restrict__ wL, const float* __restrict__ bL,
                        const float* __restrict__ fcW, const float* __restrict__ fcb,
                        const float* __restrict__ wprj, float* __restrict__ out, int n)
{
    int i = blockIdx.x * blockDim.x + threadIdx.x;
    if (i >= n * 3) return;
    const int row = i / 3, c = i % 3;
    const float* h = hgn + (long long)row * 128;
    float s1 = 0.0f, s2 = 0.0f, seq = 0.0f;
    const float ac = acon[row];
    for (int j = 0; j < 128; ++j) {
        s1  += h[j] * fcW[j * 3 + c];
        s2  += (ac * wL[j] + bL[j]) * fcW[(128 + j) * 3 + c];
        seq += h[j] * wprj[c * 128 + j];
    }
    out[(long long)row * 3 + c] = seq;                                  // seq_logit
    out[(long long)n * 3 + (long long)row * 3 + c] = s1 + s2 + fcb[c];  // pred
}

// ======================= host side =======================
struct Arena {
    char* p; size_t off;
    void* take(size_t bytes) { off = (off + 255) & ~(size_t)255; void* r = p + off; off += bytes; return r; }
};
static inline dim3 nb(long long n) { return dim3((unsigned)((n + 255) / 256)); }

static void gemmL(hipStream_t st, const bf16* A, long long sA,
                  const bf16* B, long long sB,
                  float* Cf, int ldc, long long sC, bf16* Cb, int cbMode, int cbP1, int cbP2,
                  const float* bias, int biasMask, float biasScale,
                  int M, int N, int K, int act, int acc, int batch)
{
    dim3 grid(N / 64, (M / 32 + 7) / 8, batch);
    gemm_wmma_bf16<<<grid, 256, 0, st>>>(A, sA, B, sB, Cf, Cb, ldc, sC,
                                         bias, biasMask, biasScale, M, N, K, act, acc,
                                         cbMode, cbP1, cbP2);
}
static void stageL(hipStream_t st, const float* src, int R, int C, int trans,
                   int layoutB, bf16* dst, int view, int colOff)
{
    stage_k<<<nb((long long)R * C), 256, 0, st>>>(src, R, C, trans, layoutB, dst, view, colOff);
}

extern "C" void kernel_launch(void* const* d_in, const int* in_sizes, int n_in,
                              void* d_out, int out_size, void* d_ws, size_t ws_size,
                              hipStream_t stream)
{
    (void)in_sizes; (void)n_in; (void)out_size; (void)ws_size;
    const float* x     = (const float*)d_in[0];
    const float* H     = (const float*)d_in[1];
    const float* adj   = (const float*)d_in[2];
    const float* W_gc1 = (const float*)d_in[4];
    const float* b_gc1 = (const float*)d_in[5];
    const float* W_gc2 = (const float*)d_in[6];
    const float* b_gc2 = (const float*)d_in[7];
    const float* W_m   = (const float*)d_in[8];
    const float* b_m   = (const float*)d_in[9];
    const float* W_lin = (const float*)d_in[10];
    const float* b_lin = (const float*)d_in[11];
    const float* W_ih1 = (const float*)d_in[12];
    const float* W_hh1 = (const float*)d_in[13];
    const float* b_ih1 = (const float*)d_in[14];
    const float* b_hh1 = (const float*)d_in[15];
    const float* W_ih2 = (const float*)d_in[16];
    const float* W_hh2 = (const float*)d_in[17];
    const float* b_ih2 = (const float*)d_in[18];
    const float* b_hh2 = (const float*)d_in[19];
    const float* W_q   = (const float*)d_in[20];
    const float* W_k   = (const float*)d_in[21];
    const float* W_v   = (const float*)d_in[22];
    const float* W_o   = (const float*)d_in[23];
    const float* ln_g  = (const float*)d_in[24];
    const float* ln_b  = (const float*)d_in[25];
    const float* W_h1  = (const float*)d_in[26];
    const float* b_h1  = (const float*)d_in[27];
    const float* W_h2  = (const float*)d_in[28];
    const float* b_h2  = (const float*)d_in[29];
    const float* av_w  = (const float*)d_in[30];
    const float* av_b  = (const float*)d_in[31];
    const float* av_u  = (const float*)d_in[32];
    const float* w_L   = (const float*)d_in[33];
    const float* b_L   = (const float*)d_in[34];
    const float* fc_W  = (const float*)d_in[35];
    const float* fc_b  = (const float*)d_in[36];
    const float* W_prj = (const float*)d_in[37];
    float* out = (float*)d_out;

    Arena ar{(char*)d_ws, 0};
    bf16* adjb  = (bf16*)ar.take((size_t)NNODE*NNODE*2);
    bf16* Hb    = (bf16*)ar.take((size_t)NNODE*MEDGE*2);
    bf16* HTb   = (bf16*)ar.take((size_t)MEDGE*NNODE*2);
    bf16* Hnb   = (bf16*)ar.take((size_t)NNODE*NNODE*2);
    bf16* Gb    = (bf16*)ar.take((size_t)NNODE*NNODE*2);
    bf16* wmT   = (bf16*)ar.take((size_t)MEDGE*NNODE*2);
    bf16* wgc1b = (bf16*)ar.take((size_t)FF*NHID*2);
    bf16* wgc2b = (bf16*)ar.take((size_t)NHID*FF*2);
    bf16* wlinT = (bf16*)ar.take((size_t)FF*128*2);
    bf16* wih1T = (bf16*)ar.take((size_t)128*384*2);
    bf16* whh1T = (bf16*)ar.take((size_t)128*384*2);
    bf16* wih2T = (bf16*)ar.take((size_t)128*384*2);
    bf16* whh2T = (bf16*)ar.take((size_t)128*384*2);
    bf16* wqkvT = (bf16*)ar.take((size_t)128*384*2);
    bf16* woT   = (bf16*)ar.take((size_t)128*128*2);
    bf16* wh1b  = (bf16*)ar.take((size_t)2048*128*2);
    bf16* wh2b  = (bf16*)ar.take((size_t)128*128*2);
    bf16* avwb  = (bf16*)ar.take((size_t)128*128*2);
    bf16* xb    = (bf16*)ar.take((size_t)BNT_*64*2);
    bf16* xw1b  = (bf16*)ar.take((size_t)BNT_*128*2);
    bf16* hb    = (bf16*)ar.take((size_t)BNT_*128*2);
    bf16* hw2b  = (bf16*)ar.take((size_t)BNT_*64*2);
    float* zf   = (float*)ar.take((size_t)BNT_*64*4);
    bf16* zbb   = (bf16*)ar.take((size_t)BNT_*64*2);
    bf16* xlinb = (bf16*)ar.take((size_t)BNT_*128*2);
    float* BIGf = (float*)ar.take((size_t)BNT_*384*4);   // gx L1 / gx L2 / qkv
    float* ghf  = (float*)ar.take((size_t)BN_*384*4);
    float* hsf  = (float*)ar.take((size_t)BN_*128*4);
    bf16* hswz  = (bf16*)ar.take((size_t)BN_*128*2);
    bf16* r1b   = (bf16*)ar.take((size_t)BNT_*128*2);
    bf16* rnnb  = (bf16*)ar.take((size_t)BNT_*128*2);
    float* rnnf = (float*)ar.take((size_t)BNT_*128*4);
    bf16* ob    = (bf16*)ar.take((size_t)BNT_*128*2);
    float* oprj = (float*)ar.take((size_t)BNT_*128*4);
    bf16* encb  = (bf16*)ar.take((size_t)BNT_*128*2);
    bf16* t1b   = (bf16*)ar.take((size_t)BN_*128*2);
    bf16* h1b   = (bf16*)ar.take((size_t)BN_*128*2);
    bf16* t2b   = (bf16*)ar.take((size_t)BN_*128*2);
    float* hgnf = (float*)ar.take((size_t)BN_*128*4);
    bf16* hgnb  = (bf16*)ar.take((size_t)BN_*128*2);
    float* alat = (float*)ar.take((size_t)BN_*128*4);
    float* logt = (float*)ar.take((size_t)BN_*4);
    float* alph = (float*)ar.take((size_t)BN_*4);
    float* acon = (float*)ar.take((size_t)BN_*4);

    // ---- stage operands into fragment-swizzled bf16 ----
    stageL(stream, adj,   NNODE, NNODE, 0, 0, adjb, NNODE, 0);
    stageL(stream, H,     NNODE, MEDGE, 0, 0, Hb,   MEDGE, 0);
    stageL(stream, H,     MEDGE, NNODE, 1, 1, HTb,  NNODE, 0);   // H^T as B
    stageL(stream, W_m,   MEDGE, NNODE, 1, 1, wmT,  NNODE, 0);   // W_m^T as B
    stageL(stream, W_gc1, FF, NHID, 0, 1, wgc1b, NHID, 0);
    stageL(stream, W_gc2, NHID, FF, 0, 1, wgc2b, FF, 0);
    stageL(stream, W_lin, FF, 128, 1, 1, wlinT, 128, 0);
    stageL(stream, W_ih1, 128, 384, 1, 1, wih1T, 384, 0);
    stageL(stream, W_hh1, 128, 384, 1, 1, whh1T, 384, 0);
    stageL(stream, W_ih2, 128, 384, 1, 1, wih2T, 384, 0);
    stageL(stream, W_hh2, 128, 384, 1, 1, whh2T, 384, 0);
    stageL(stream, W_q,   128, 128, 1, 1, wqkvT, 384, 0);
    stageL(stream, W_k,   128, 128, 1, 1, wqkvT, 384, 128);
    stageL(stream, W_v,   128, 128, 1, 1, wqkvT, 384, 256);
    stageL(stream, W_o,   128, 128, 1, 1, woT, 128, 0);
    stageL(stream, W_h1,  2048, 128, 0, 1, wh1b, 128, 0);
    stageL(stream, W_h2,  128, 128, 0, 1, wh2b, 128, 0);
    stageL(stream, av_w,  128, 128, 0, 1, avwb, 128, 0);
    stageL(stream, x,     BNT_, 64, 0, 0, xb, 64, 0);

    // Hn = H @ W_m^T + b_m (as A operand); G = H @ H^T (as A operand)
    gemmL(stream, Hb, 0, wmT, 0, nullptr, 0, 0, Hnb, 1, NNODE, 0,
          b_m, 1023, 1.f, NNODE, NNODE, MEDGE, 0, 0, 1);
    gemmL(stream, Hb, 0, HTb, 0, nullptr, 0, 0, Gb, 1, NNODE, 0,
          nullptr, 0, 0.f, NNODE, NNODE, MEDGE, 0, 0, 1);

    // ---- GCN ----
    const long long sTH = (long long)NNODE * 2048, sTF = (long long)NNODE * 1024;
    gemmL(stream, xb, 0, wgc1b, 0, nullptr, 0, 0, xw1b, 4, 2048, 7,
          nullptr, 0, 0.f, BNT_, 128, 64, 0, 0, 1);
    // path 1: adj
    gemmL(stream, adjb, 0, xw1b, sTH, nullptr, 0, sTH, hb, 3, 0, 7,
          b_gc1, 127, 1.f, NNODE, 2048, NNODE, 1, 0, BB);
    gemmL(stream, hb, 0, wgc2b, 0, nullptr, 0, 0, hw2b, 4, 1024, 6,
          nullptr, 0, 0.f, BNT_, 64, 128, 0, 0, 1);
    gemmL(stream, adjb, 0, hw2b, sTF, zf, 1024, sTF, nullptr, 0, 0, 0,
          nullptr, 0, 0.f, NNODE, 1024, NNODE, 0, 0, BB);
    // path 2: Hn (accumulate; both _gcn calls add b_gc2 -> scale 2)
    gemmL(stream, Hnb, 0, xw1b, sTH, nullptr, 0, sTH, hb, 3, 0, 7,
          b_gc1, 127, 1.f, NNODE, 2048, NNODE, 1, 0, BB);
    gemmL(stream, hb, 0, wgc2b, 0, nullptr, 0, 0, hw2b, 4, 1024, 6,
          nullptr, 0, 0.f, BNT_, 64, 128, 0, 0, 1);
    gemmL(stream, Hnb, 0, hw2b, sTF, zf, 1024, sTF, zbb, 3, 0, 6,
          b_gc2, 63, 2.f, NNODE, 1024, NNODE, 0, 1, BB);

    // x_lin = z @ W_lin^T + b_lin
    gemmL(stream, zbb, 0, wlinT, 0, nullptr, 0, 0, xlinb, 1, 128, 0,
          b_lin, 127, 1.f, BNT_, 128, 64, 0, 0, 1);

    // ---- GRU layer 1 ----
    gemmL(stream, xlinb, 0, wih1T, 0, BIGf, 384, 0, nullptr, 0, 0, 0,
          b_ih1, 511, 1.f, BNT_, 384, 128, 0, 0, 1);
    for (int t = 0; t < TT; ++t) {
        if (t > 0)
            gemmL(stream, hswz, 0, whh1T, 0, ghf, 384, 0, nullptr, 0, 0, 0,
                  b_hh1, 511, 1.f, BN_, 384, 128, 0, 0, 1);
        gru_gate_k<<<nb((long long)BN_*128), 256, 0, stream>>>(
            BIGf, t ? ghf : nullptr, b_hh1, hsf, hswz, r1b, nullptr, t, BN_*128);
    }
    // ---- GRU layer 2 ----
    gemmL(stream, r1b, 0, wih2T, 0, BIGf, 384, 0, nullptr, 0, 0, 0,
          b_ih2, 511, 1.f, BNT_, 384, 128, 0, 0, 1);
    for (int t = 0; t < TT; ++t) {
        if (t > 0)
            gemmL(stream, hswz, 0, whh2T, 0, ghf, 384, 0, nullptr, 0, 0, 0,
                  b_hh2, 511, 1.f, BN_, 384, 128, 0, 0, 1);
        gru_gate_k<<<nb((long long)BN_*128), 256, 0, stream>>>(
            BIGf, t ? ghf : nullptr, b_hh2, hsf, hswz, rnnb, rnnf, t, BN_*128);
    }

    // ---- attention ----
    gemmL(stream, rnnb, 0, wqkvT, 0, BIGf, 384, 0, nullptr, 0, 0, 0,
          nullptr, 0, 0.f, BNT_, 384, 128, 0, 0, 1);
    attn_core_k<<<nb((long long)BN_*4*TT), 256, 0, stream>>>(BIGf, ob, BN_*4*TT);
    gemmL(stream, ob, 0, woT, 0, oprj, 128, 0, nullptr, 0, 0, 0,
          nullptr, 0, 0.f, BNT_, 128, 128, 0, 0, 1);
    ln_res_k<<<nb(BNT_), 256, 0, stream>>>(rnnf, oprj, ln_g, ln_b, encb, BNT_);

    // ---- HGN_AD ----
    const long long sNH = (long long)NNODE * 128;
    gemmL(stream, encb, 0, wh1b, 0, nullptr, 0, 0, t1b, 5, 128, 0,
          nullptr, 0, 0.f, BN_, 128, 2048, 0, 0, 1);
    gemmL(stream, Gb, 0, t1b, sNH, nullptr, 0, sNH, h1b, 1, 128, 0,
          b_h1, 127, 1.f, NNODE, 128, NNODE, 1, 0, BB);
    gemmL(stream, h1b, 0, wh2b, 0, nullptr, 0, 0, t2b, 5, 128, 0,
          nullptr, 0, 0.f, BN_, 128, 128, 0, 0, 1);
    gemmL(stream, adjb, 0, t2b, sNH, hgnf, 128, sNH, hgnb, 1, 128, 0,
          b_h2, 127, 1.f, NNODE, 128, NNODE, 0, 0, BB);

    // construct_graph + heads
    gemmL(stream, hgnb, 0, avwb, 0, alat, 128, 0, nullptr, 0, 0, 0,
          av_b, 127, 1.f, BN_, 128, 128, 2, 0, 1);
    rowdot_k<<<nb(BN_), 256, 0, stream>>>(alat, av_u, logt, BN_, 128);
    softmax_bn_k<<<1, 256, 0, stream>>>(logt, alph, BN_);
    acon_k<<<nb(BN_), 256, 0, stream>>>(hgnf, alph, acon, BN_);
    heads_k<<<nb((long long)BN_*3), 256, 0, stream>>>(hgnf, acon, w_L, b_L,
                                                      fc_W, fc_b, W_prj, out, BN_);
}
